// SandayASR_87110526698093
// MI455X (gfx1250) — matrix operand order repaired and checked by reading, not compiled, not run on hardware
//
#include <hip/hip_runtime.h>
#include <hip/hip_bf16.h>
#include <math.h>

// ---------------------------------------------------------------------------
// Model dims
#define B_    16
#define T_    800
#define D_    80
#define H_    256
#define R_    16
#define P_    45
#define PPAD  48
#define D0PAD 128   // layer-0 input dim padded to a multiple of 32 (K for WMMA)

typedef __bf16 bf16_t;
typedef __attribute__((ext_vector_type(16))) __bf16    v16bf;
typedef __attribute__((ext_vector_type(8)))  float     v8f;
typedef __attribute__((ext_vector_type(2)))  _Float16  f16x2;

union V16U { v16bf v; uint4 q[2]; };

// Prevent LICM from hoisting weight loads out of the time loop (keeps the
// per-step global_load_b128 stream to L2 instead of spilling a private copy
// of all fragments to scratch).  "s" keeps the base in SGPRs => saddr form.
__device__ __forceinline__ const bf16_t* opq(const bf16_t* p) {
    asm volatile("" : "+s"(p));
    return p;
}

// Branch-free tanh: 1 - 2/(1+exp(2x)); v_exp_f32 + v_rcp_f32, saturates
// correctly at +/-1.  Keeps transient register pressure minimal.
__device__ __forceinline__ float fast_tanh(float x) {
    float e = __builtin_amdgcn_exp2f(x * 2.8853900817779268f);  // exp(2x)
    return 1.0f - 2.0f * __builtin_amdgcn_rcpf(1.0f + e);
}

// ---------------------------------------------------------------------------
// WMMA fragment loaders.
// A-operand (16xK bf16, row-major source in LDS):
//   lane l: m = l&15 ; lanes 0-15 hold K = {kc*32+0..7, kc*32+16..23},
//                      lanes 16-31 hold K = {kc*32+8..15, kc*32+24..31}
__device__ __forceinline__ v16bf load_afrag(const bf16_t* rowbase, int kc, int hi) {
    V16U u;
    const bf16_t* p = rowbase + kc * 32 + hi * 8;
    u.q[0] = *(const uint4*)(p);
    u.q[1] = *(const uint4*)(p + 16);
    return u.v;
}
// B-operand: pre-packed lane-major fragments; 32 contiguous bytes per lane.
__device__ __forceinline__ v16bf load_bfrag(const bf16_t* fragbase) {
    V16U u;
    const uint4* p = (const uint4*)fragbase;
    u.q[0] = p[0];
    u.q[1] = p[1];
    return u.v;
}
__device__ __forceinline__ v8f wmma_bf16(v16bf a, v16bf b, v8f c) {
    return __builtin_amdgcn_wmma_f32_16x16x32_bf16(false, a, false, b,
                                                   (short)0, c, false, false);
}

// ---------------------------------------------------------------------------
// Weight packing: f32 row-major (K x N) -> bf16 B-operand fragments,
// zero-padded to (Kpad x Npad). Fragment f = nt*(Kpad/32)+kc, element index
// = (f*32 + lane)*16 + e ;  col = nt*16 + (lane&15),
// k = kc*32 + (lane>=16 ? 8 : 0) + (e&7) + (e>=8 ? 16 : 0).
__global__ void pack_bfrag_kernel(const float* __restrict__ src,
                                  bf16_t* __restrict__ dst,
                                  int K, int N, int Kpad, int Npad) {
    int nch = Kpad >> 5;
    int total = Kpad * Npad;
    for (int i = blockIdx.x * blockDim.x + threadIdx.x; i < total;
         i += gridDim.x * blockDim.x) {
        int e    = i & 15;
        int lane = (i >> 4) & 31;
        int f    = i >> 9;
        int kc   = f % nch;
        int nt   = f / nch;
        int n    = nt * 16 + (lane & 15);
        int k    = kc * 32 + ((lane >> 4) << 3) + (e & 7) + ((e >> 3) << 4);
        float v  = (k < K && n < N) ? src[k * N + n] : 0.0f;
        dst[i] = (bf16_t)v;
    }
}

// ---------------------------------------------------------------------------
// Persistent single-workgroup DREAM recurrence.
// 256 threads = 8 waves (2 waves/SIMD => large VGPR budget); wave w owns
// output columns [32w, 32w+32) (two 16-col WMMA N-tiles) of every layer.
// Fast-weight state A is register resident as packed f16 pairs:
// Areg[L][tile][b_lo][r-pair], lane = (b_hi*16 + col_in_tile) -- matches the
// WMMA D-fragment layout, so readout/update are lane-local packed-f16 FMAs
// (v_pk_fma_f16) co-executing with the WMMA chain.  Readout, tanh, h-store
// and the Hebbian update are fused per b_lo so each k-row and A element is
// touched exactly once per step (minimum peak liveness).
__global__ __launch_bounds__(256, 1)
void dream_forward(const float* __restrict__ x,
                   const bf16_t* __restrict__ wi0, const bf16_t* __restrict__ wh0,
                   const bf16_t* __restrict__ wk0,
                   const bf16_t* __restrict__ wi1, const bf16_t* __restrict__ wh1,
                   const bf16_t* __restrict__ wk1,
                   const bf16_t* __restrict__ wi2, const bf16_t* __restrict__ wh2,
                   const bf16_t* __restrict__ wk2,
                   const float* __restrict__ b0, const float* __restrict__ b1,
                   const float* __restrict__ b2,
                   float* __restrict__ y) {
    __shared__ alignas(16) bf16_t   hbuf[4][16][H_];  // [0]=x_t(pad), [1..3]=h(layer)
    __shared__ alignas(16) _Float16 klds[B_ * R_];    // k_t for current layer, [b][r]

    const int tid  = threadIdx.x;
    const int wave = tid >> 5;          // 0..7
    const int lane = tid & 31;
    const int nlo  = lane & 15;
    const int hi   = lane >> 4;
    const int col0 = wave * 32 + nlo;   // N-tile 2w
    const int col1 = col0 + 16;         // N-tile 2w+1

    const float biasr[3][2] = { { b0[col0], b0[col1] },
                                { b1[col0], b1[col1] },
                                { b2[col0], b2[col1] } };
    const float invt[3]   = { 0.5f, 0.25f, 0.125f };   // 1/tau, tau = 2,4,8
    const float decayc[3] = { 0.5f, 0.75f, 0.875f };

    // Zero h state and the x padding columns once.
    for (int i = tid; i < 4 * 16 * H_; i += 256)
        ((bf16_t*)hbuf)[i] = (bf16_t)0.0f;

    // Fast weights A: 3 layers x 2 tiles x 8 b_lo x 8 r-pairs, one VGPR per
    // f16 pair => 384 VGPRs/lane (uses gfx1250 high-VGPR addressing).
    f16x2 Areg[3][2][8][8];
#pragma unroll
    for (int L = 0; L < 3; ++L)
#pragma unroll
        for (int tl = 0; tl < 2; ++tl)
#pragma unroll
            for (int v = 0; v < 8; ++v)
#pragma unroll
                for (int rp = 0; rp < 8; ++rp) {
                    Areg[L][tl][v][rp][0] = (_Float16)0.0f;
                    Areg[L][tl][v][rp][1] = (_Float16)0.0f;
                }

    __syncthreads();

    for (int t = 0; t < T_; ++t) {
        // Re-launder weight bases each step: blocks load hoisting / spilling.
        const bf16_t* wiP[3] = { opq(wi0), opq(wi1), opq(wi2) };
        const bf16_t* whP[3] = { opq(wh0), opq(wh1), opq(wh2) };
        const bf16_t* wkP[3] = { opq(wk0), opq(wk1), opq(wk2) };

        // Stage x[:, t, :] into hbuf[0] (cols 80..127 stay zero).
        for (int i = tid; i < B_ * D_; i += 256) {
            int b = i / D_, d = i - b * D_;
            hbuf[0][b][d] = (bf16_t)x[(b * T_ + t) * D_ + d];
        }
        __syncthreads();

#pragma unroll
        for (int L = 0; L < 3; ++L) {
            const int CIN = (L == 0) ? (D0PAD / 32) : (H_ / 32);
            const bf16_t* arow_cur = &hbuf[L][nlo][0];
            const bf16_t* arow_hp  = &hbuf[L + 1][nlo][0];
            const int nt0 = 2 * wave, nt1 = 2 * wave + 1;

            // ---- GEMM: acc = cur @ W_in + h_prev @ W_h  (two 16-col tiles)
            v8f acc0 = { 0, 0, 0, 0, 0, 0, 0, 0 };
            v8f acc1 = { 0, 0, 0, 0, 0, 0, 0, 0 };
#pragma unroll
            for (int kc = 0; kc < CIN; ++kc) {
                v16bf a   = load_afrag(arow_cur, kc, hi);
                v16bf bm0 = load_bfrag(wiP[L] + ((nt0 * CIN + kc) * 32 + lane) * 16);
                v16bf bm1 = load_bfrag(wiP[L] + ((nt1 * CIN + kc) * 32 + lane) * 16);
                acc0 = wmma_bf16(a, bm0, acc0);
                acc1 = wmma_bf16(a, bm1, acc1);
            }
#pragma unroll
            for (int kc = 0; kc < 8; ++kc) {
                v16bf a   = load_afrag(arow_hp, kc, hi);
                v16bf bm0 = load_bfrag(whP[L] + ((nt0 * 8 + kc) * 32 + lane) * 16);
                v16bf bm1 = load_bfrag(whP[L] + ((nt1 * 8 + kc) * 32 + lane) * 16);
                acc0 = wmma_bf16(a, bm0, acc0);
                acc1 = wmma_bf16(a, bm1, acc1);
            }

            // ---- wave 0: k_t = cur @ W_k (16x16) via WMMA, broadcast via LDS
            if (wave == 0) {
                v8f kacc = { 0, 0, 0, 0, 0, 0, 0, 0 };
#pragma unroll
                for (int kc = 0; kc < CIN; ++kc) {
                    v16bf a  = load_afrag(arow_cur, kc, hi);
                    v16bf bm = load_bfrag(wkP[L] + (kc * 32 + lane) * 16);
                    kacc = wmma_bf16(a, bm, kacc);
                }
#pragma unroll
                for (int v = 0; v < 8; ++v)
                    klds[(hi * 8 + v) * R_ + nlo] = (_Float16)kacc[v];
            }
            __syncthreads();   // k ready; all GEMM reads of hbuf complete

            // ---- fused: readout -> tanh -> h store -> Hebbian update.
            //   read[b]   = sum_r k[b][r] * A[b][r][col]
            //   h         = tanh(gemm + read + bias)
            //   A         = decay*A + (1/tau) * k (x) h
            _Float16 dh = (_Float16)decayc[L];
            f16x2 d2 = { dh, dh };
#pragma unroll
            for (int v = 0; v < 8; ++v) {
                int b = hi * 8 + v;
                const f16x2* kp = (const f16x2*)&klds[b * R_];
                f16x2 kk[8];
#pragma unroll
                for (int rp = 0; rp < 8; ++rp) kk[rp] = kp[rp];

                f16x2 r0 = { (_Float16)0.0f, (_Float16)0.0f };
                f16x2 r1 = { (_Float16)0.0f, (_Float16)0.0f };
#pragma unroll
                for (int rp = 0; rp < 8; ++rp) {
                    r0 = kk[rp] * Areg[L][0][v][rp] + r0;   // v_pk_fma_f16
                    r1 = kk[rp] * Areg[L][1][v][rp] + r1;
                }
                float hv0 = fast_tanh(acc0[v] + (float)r0[0] + (float)r0[1] + biasr[L][0]);
                float hv1 = fast_tanh(acc1[v] + (float)r1[0] + (float)r1[1] + biasr[L][1]);

                hbuf[L + 1][b][col0] = (bf16_t)hv0;
                hbuf[L + 1][b][col1] = (bf16_t)hv1;
                if (L == 2) {
                    y[((size_t)b * T_ + t) * H_ + col0] = hv0;
                    y[((size_t)b * T_ + t) * H_ + col1] = hv1;
                }

                _Float16 c0 = (_Float16)(invt[L] * hv0);
                _Float16 c1 = (_Float16)(invt[L] * hv1);
                f16x2 c20 = { c0, c0 };
                f16x2 c21 = { c1, c1 };
#pragma unroll
                for (int rp = 0; rp < 8; ++rp) {
                    Areg[L][0][v][rp] = Areg[L][0][v][rp] * d2 + kk[rp] * c20;
                    Areg[L][1][v][rp] = Areg[L][1][v][rp] * d2 + kk[rp] * c21;
                }
            }
            __syncthreads();   // h_new visible to next layer
        }
    }
}

// ---------------------------------------------------------------------------
// CTC head: logits = hidden(12800x256) @ Wc(256x48pad) + bc, via WMMA.
// One block per M-tile (16 rows); 3 waves = 3 N-tiles (48 padded cols).
__global__ __launch_bounds__(96)
void ctc_head(const float* __restrict__ y, const bf16_t* __restrict__ wcf,
              const float* __restrict__ bc, float* __restrict__ outLogits) {
    int mt   = blockIdx.x;
    int wave = threadIdx.x >> 5;   // N-tile 0..2
    int lane = threadIdx.x & 31;
    int nlo  = lane & 15, hi = lane >> 4;

    v8f acc = { 0, 0, 0, 0, 0, 0, 0, 0 };
#pragma unroll
    for (int kc = 0; kc < 8; ++kc) {
        const float* rp = y + (size_t)(mt * 16 + nlo) * H_ + kc * 32 + hi * 8;
        float4 f0 = *(const float4*)(rp);
        float4 f1 = *(const float4*)(rp + 4);
        float4 f2 = *(const float4*)(rp + 16);
        float4 f3 = *(const float4*)(rp + 20);
        v16bf a;
        a[0]  = (bf16_t)f0.x; a[1]  = (bf16_t)f0.y; a[2]  = (bf16_t)f0.z; a[3]  = (bf16_t)f0.w;
        a[4]  = (bf16_t)f1.x; a[5]  = (bf16_t)f1.y; a[6]  = (bf16_t)f1.z; a[7]  = (bf16_t)f1.w;
        a[8]  = (bf16_t)f2.x; a[9]  = (bf16_t)f2.y; a[10] = (bf16_t)f2.z; a[11] = (bf16_t)f2.w;
        a[12] = (bf16_t)f3.x; a[13] = (bf16_t)f3.y; a[14] = (bf16_t)f3.z; a[15] = (bf16_t)f3.w;
        v16bf bm = load_bfrag(wcf + ((wave * 8 + kc) * 32 + lane) * 16);
        acc = wmma_bf16(a, bm, acc);
    }
    int p = wave * 16 + nlo;
    float bcv = (p < P_) ? bc[p] : 0.0f;
#pragma unroll
    for (int v = 0; v < 8; ++v) {
        int row = mt * 16 + hi * 8 + v;
        if (p < P_)
            outLogits[(size_t)row * P_ + p] = acc[v] + bcv;
    }
}

// ---------------------------------------------------------------------------
// Row-wise log-softmax over P=45, one wave32 per row.
__global__ __launch_bounds__(256)
void log_softmax_k(const float* __restrict__ logits, float* __restrict__ lp) {
    int w    = (int)((blockIdx.x * blockDim.x + threadIdx.x) >> 5);  // row
    int lane = threadIdx.x & 31;
    const float* rowp = logits + (size_t)w * P_;
    float v0 = (lane < P_)      ? rowp[lane]      : -3.0e38f;
    float v1 = (lane + 32 < P_) ? rowp[lane + 32] : -3.0e38f;
    float m = fmaxf(v0, v1);
#pragma unroll
    for (int off = 16; off >= 1; off >>= 1)
        m = fmaxf(m, __shfl_xor(m, off, 32));
    float s = 0.0f;
    if (lane < P_)      s += expf(v0 - m);
    if (lane + 32 < P_) s += expf(v1 - m);
#pragma unroll
    for (int off = 16; off >= 1; off >>= 1)
        s += __shfl_xor(s, off, 32);
    float lse = logf(s);
    if (lane < P_)      lp[(size_t)w * P_ + lane]      = v0 - m - lse;
    if (lane + 32 < P_) lp[(size_t)w * P_ + lane + 32] = v1 - m - lse;
}

// ---------------------------------------------------------------------------
extern "C" void kernel_launch(void* const* d_in, const int* in_sizes, int n_in,
                              void* d_out, int out_size, void* d_ws, size_t ws_size,
                              hipStream_t stream) {
    (void)in_sizes; (void)n_in; (void)out_size; (void)ws_size;

    const float* x   = (const float*)d_in[0];
    const float* Wi0 = (const float*)d_in[1];
    const float* Wh0 = (const float*)d_in[2];
    const float* Wk0 = (const float*)d_in[3];
    const float* b0  = (const float*)d_in[4];
    const float* Wi1 = (const float*)d_in[5];
    const float* Wh1 = (const float*)d_in[6];
    const float* Wk1 = (const float*)d_in[7];
    const float* b1  = (const float*)d_in[8];
    const float* Wi2 = (const float*)d_in[9];
    const float* Wh2 = (const float*)d_in[10];
    const float* Wk2 = (const float*)d_in[11];
    const float* b2  = (const float*)d_in[12];
    const float* Wc  = (const float*)d_in[13];
    const float* bc  = (const float*)d_in[14];

    char*  ws  = (char*)d_ws;
    float* y   = (float*)ws;
    size_t off = (size_t)B_ * T_ * H_ * sizeof(float);
    auto alloc = [&](size_t elems) -> bf16_t* {
        bf16_t* p = (bf16_t*)(ws + off);
        off += elems * sizeof(bf16_t);
        off = (off + 255) & ~(size_t)255;
        return p;
    };
    bf16_t* wi0f = alloc((size_t)D0PAD * H_);
    bf16_t* wh0f = alloc((size_t)H_ * H_);
    bf16_t* wk0f = alloc((size_t)D0PAD * R_);
    bf16_t* wi1f = alloc((size_t)H_ * H_);
    bf16_t* wh1f = alloc((size_t)H_ * H_);
    bf16_t* wk1f = alloc((size_t)H_ * R_);
    bf16_t* wi2f = alloc((size_t)H_ * H_);
    bf16_t* wh2f = alloc((size_t)H_ * H_);
    bf16_t* wk2f = alloc((size_t)H_ * R_);
    bf16_t* wcf  = alloc((size_t)H_ * PPAD);

    pack_bfrag_kernel<<<64, 256, 0, stream>>>(Wi0, wi0f, D_, H_, D0PAD, H_);
    pack_bfrag_kernel<<<64, 256, 0, stream>>>(Wh0, wh0f, H_, H_, H_, H_);
    pack_bfrag_kernel<<<64, 256, 0, stream>>>(Wk0, wk0f, D_, R_, D0PAD, R_);
    pack_bfrag_kernel<<<64, 256, 0, stream>>>(Wi1, wi1f, H_, H_, H_, H_);
    pack_bfrag_kernel<<<64, 256, 0, stream>>>(Wh1, wh1f, H_, H_, H_, H_);
    pack_bfrag_kernel<<<64, 256, 0, stream>>>(Wk1, wk1f, H_, R_, H_, R_);
    pack_bfrag_kernel<<<64, 256, 0, stream>>>(Wi2, wi2f, H_, H_, H_, H_);
    pack_bfrag_kernel<<<64, 256, 0, stream>>>(Wh2, wh2f, H_, H_, H_, H_);
    pack_bfrag_kernel<<<64, 256, 0, stream>>>(Wk2, wk2f, H_, R_, H_, R_);
    pack_bfrag_kernel<<<64, 256, 0, stream>>>(Wc,  wcf,  H_, P_, H_, PPAD);

    dream_forward<<<1, 256, 0, stream>>>(x,
                                         wi0f, wh0f, wk0f,
                                         wi1f, wh1f, wk1f,
                                         wi2f, wh2f, wk2f,
                                         b0, b1, b2, y);

    float* outLogits = (float*)d_out;
    float* outLP     = outLogits + (size_t)B_ * T_ * P_;
    ctc_head<<<(B_ * T_) / 16, 96, 0, stream>>>(y, wcf, bc, outLogits);
    log_softmax_k<<<(B_ * T_) / 8, 256, 0, stream>>>(outLogits, outLP);
}
